// LSTM_Network_57088705299037
// MI455X (gfx1250) — compile-verified, matrix operand order checked by reading, not run on hardware
//
#include <hip/hip_runtime.h>

#define B_DIM 512
#define T_DIM 256
#define NIN   128
#define H_DIM 1024

typedef __attribute__((ext_vector_type(16))) __bf16        v16bf;
typedef __attribute__((ext_vector_type(8)))  float         v8f;
typedef __attribute__((ext_vector_type(4)))  unsigned int  v4u;
typedef __attribute__((ext_vector_type(4)))  int           v4i;

union FragBF { v16bf v; v4u q[2]; };

// ---- CDNA5 async global->LDS path (guarded; falls back to load+ds_store) ----
#if defined(__HIP_DEVICE_COMPILE__) && defined(__has_builtin)
#  if __has_builtin(__builtin_amdgcn_global_load_async_to_lds_b128)
#    define USE_ASYNC_LDS 1
#  endif
#endif

__device__ __forceinline__ void stage16(const unsigned short* g, unsigned short* l) {
#if defined(USE_ASYNC_LDS)
  // (int4* global src, int4* LDS dst, imm offset, imm cpol); clang inserts AS casts.
  __builtin_amdgcn_global_load_async_to_lds_b128((v4i*)g, (v4i*)l, 0, 0);
#else
  *(v4u*)l = *(const v4u*)g;
#endif
}

__device__ __forceinline__ void stage_wait() {
#if defined(USE_ASYNC_LDS)
#  if __has_builtin(__builtin_amdgcn_s_wait_asynccnt)
  __builtin_amdgcn_s_wait_asynccnt(0);
#  else
  asm volatile("s_wait_asynccnt 0x0" ::: "memory");
#  endif
#endif
}

__device__ __forceinline__ unsigned short f2bf(float x) {
  unsigned int u = __float_as_uint(x);
  u += 0x7FFFu + ((u >> 16) & 1u);          // round-to-nearest-even
  return (unsigned short)(u >> 16);
}
__device__ __forceinline__ float fsig(float x)  { return 1.0f / (1.0f + __expf(-x)); }
__device__ __forceinline__ float ftanh(float x) { return 2.0f / (1.0f + __expf(-2.0f * x)) - 1.0f; }

// Load a 16x32 bf16 WMMA operand fragment (works for global or LDS pointers).
// Per ISA 7.12.2: lane owns row/col (lane&15); lanes 16..31 take k offsets +8/+24,
// i.e. two contiguous 16B chunks at +0 and +32 bytes from (base + (lane>>4)*16B).
__device__ __forceinline__ v16bf load_frag(const unsigned short* p0, int lane) {
  const unsigned short* p = p0 + ((lane >> 4) << 3);
  FragBF f;
  f.q[0] = *(const v4u*)(p);
  f.q[1] = *(const v4u*)(p + 16);
  return f.v;
}

__device__ __forceinline__ v8f wmma_bf16(v16bf a, v16bf b, v8f c) {
  return __builtin_amdgcn_wmma_f32_16x16x32_bf16(false, a, false, b, (short)0, c, false, false);
}

// ---------------- conversion kernels ----------------

// dst[c*R + r] = bf16(src[r*C + c]); 2D grid, no integer division.
// grid: (C/256, R), block 256. Coalesced reads along the source row.
__global__ void convT_kernel(const float* __restrict__ src, unsigned short* __restrict__ dst,
                             int R, int C) {
  const int c = blockIdx.x * blockDim.x + threadIdx.x;
  const int r = blockIdx.y;
  if (c >= C) return;
  dst[(size_t)c * R + r] = f2bf(src[(size_t)r * C + c]);
}

__global__ void conv_kernel(const float* __restrict__ src, unsigned short* __restrict__ dst, long n) {
  long i = (long)blockIdx.x * blockDim.x + threadIdx.x;
  if (i < n) dst[i] = f2bf(src[i]);
}

// ---------------- input projection: proj[t*B+b, :] = relu(x[b,t,:] @ Wh + bh) ----
// grid: (BT/128, H/64), block 256 (8 waves, each 16 rows x 64 cols). K=128 only.
__global__ __launch_bounds__(256) void proj_kernel(
    const unsigned short* __restrict__ Xbf,   // [B*T, 128] in (b,t) order
    const unsigned short* __restrict__ WhT,   // [1024, 128] (N-major)
    const float* __restrict__ bh,             // [1024]
    unsigned short* __restrict__ proj) {      // [T*B, 1024] time-major
  const int tid = threadIdx.x, lane = tid & 31, wid = tid >> 5;
  const int nl = lane & 15;
  const long rBase = (long)blockIdx.x * 128 + wid * 16;
  const int  nBase = blockIdx.y * 64;

  const long r = rBase + nl;
  const int  t = (int)(r >> 9);
  const int  b = (int)(r & 511);
  const unsigned short* arow = Xbf + ((long)b * T_DIM + t) * NIN;

  v8f acc[4] = {};
  for (int k0 = 0; k0 < NIN; k0 += 32) {
    v16bf a = load_frag(arow + k0, lane);
#pragma unroll
    for (int g = 0; g < 4; ++g) {
      v16bf bf = load_frag(WhT + (nBase + g * 16 + nl) * NIN + k0, lane);
      acc[g] = wmma_bf16(a, bf, acc[g]);
    }
  }
  const int mHalf = (lane >> 4) << 3;
#pragma unroll
  for (int g = 0; g < 4; ++g) {
    const int n = nBase + g * 16 + nl;
    const float bb = bh[n];
#pragma unroll
    for (int e = 0; e < 8; ++e) {
      long m = rBase + mHalf + e;
      float v = acc[g][e] + bb;
      v = v > 0.0f ? v : 0.0f;
      proj[m * H_DIM + n] = f2bf(v);
    }
  }
}

// ---------------- fused LSTM step (one cell) ----------------
// z = [Ax | Ah] @ W^T + bias ; gates i,j,f,o fused into the epilogue.
// Workgroup: 256 thr = 8 waves (2 row-waves x 4 col-waves); WG tile 64 rows x 64 gate-cols.
// Wave tile: 32 rows x 16 cols x 4 gates -> 8 accumulators, 8 WMMA per 32-wide k-tile.
// A (64x32) and B (256x32, 4 gates) k-tiles are double-buffered in LDS via the
// CDNA5 async global->LDS path, staged one iteration ahead of the WMMA consume.
#define LDS_STRIDE 40   // 32 data + 8 pad ushorts (80B rows, 16B aligned, bank-spread)

__global__ __launch_bounds__(256) void lstm_step_kernel(
    const unsigned short* __restrict__ Ax,    // [512,1024] bf16 (x-part rows)
    const unsigned short* __restrict__ Ah,    // [512,1024] bf16 (h_prev)
    const unsigned short* __restrict__ Wt,    // [4096,2048] bf16 (N-major kernel)
    const float* __restrict__ bias,           // [4096]
    float* __restrict__ c,                    // [512,1024] fp32, in/out
    unsigned short* __restrict__ h_out,       // [512,1024] bf16
    float* __restrict__ out_f32,              // final-step fp32 output (or null)
    int write_out) {
  __shared__ unsigned short Ash[2][64][LDS_STRIDE];    // 10 KB
  __shared__ unsigned short Bsh[2][256][LDS_STRIDE];   // 40 KB

  const int tid = threadIdx.x, lane = tid & 31, wid = tid >> 5;
  const int nl = lane & 15;
  const int wrow = wid >> 2;                  // 0..1
  const int wcol = wid & 3;                   // 0..3
  const int mWG = blockIdx.x * 64;
  const int nWG = blockIdx.y * 64;

  // cooperative staging assignment
  const int arow = tid >> 2, achk = tid & 3;          // A: 64 rows x 4 chunks of 16B
  const int bg = tid >> 6, bcol = tid & 63;           // B: each thread one 64B row
  const unsigned short* AxRow = Ax + (size_t)(mWG + arow) * 1024 + achk * 8;
  const unsigned short* AhRow = Ah + (size_t)(mWG + arow) * 1024 + achk * 8;
  const unsigned short* WtRow = Wt + (size_t)(nWG + bcol + bg * 1024) * 2048;

#define STAGE(it)                                                              \
  {                                                                            \
    const int k0_ = (it) * 32;                                                 \
    const int sb_ = (it) & 1;                                                  \
    const unsigned short* ga_ =                                                \
        (k0_ < 1024) ? (AxRow + k0_) : (AhRow + (k0_ - 1024));                 \
    stage16(ga_, &Ash[sb_][arow][achk * 8]);                                   \
    _Pragma("unroll")                                                          \
    for (int cc_ = 0; cc_ < 4; ++cc_)                                          \
      stage16(WtRow + k0_ + cc_ * 8, &Bsh[sb_][tid][cc_ * 8]);                 \
  }

  v8f acc[2][4] = {};

  STAGE(0);
  stage_wait();
  __syncthreads();

  for (int it = 0; it < 64; ++it) {
    if (it + 1 < 64) STAGE(it + 1);           // stage next k-tile (async, in flight)
    const int sb = it & 1;
    v16bf af0 = load_frag(&Ash[sb][wrow * 32 + nl][0], lane);
    v16bf af1 = load_frag(&Ash[sb][wrow * 32 + 16 + nl][0], lane);
#pragma unroll
    for (int g = 0; g < 4; ++g) {
      v16bf bfr = load_frag(&Bsh[sb][g * 64 + wcol * 16 + nl][0], lane);
      acc[0][g] = wmma_bf16(af0, bfr, acc[0][g]);
      acc[1][g] = wmma_bf16(af1, bfr, acc[1][g]);
    }
    if (it + 1 < 64) stage_wait();
    __syncthreads();
  }

  // fused gate epilogue: i, j, f, o -> c', h'
  const int n = nWG + wcol * 16 + nl;
  const float bi = bias[n];
  const float bj = bias[n + 1024];
  const float bff = bias[n + 2048];
  const float bo = bias[n + 3072];
  const int mHalf = (lane >> 4) << 3;

#pragma unroll
  for (int mt = 0; mt < 2; ++mt) {
#pragma unroll
    for (int e = 0; e < 8; ++e) {
      const int m = mWG + wrow * 32 + mt * 16 + mHalf + e;
      const int idx = m * H_DIM + n;
      const float zi = acc[mt][0][e] + bi;
      const float zj = acc[mt][1][e] + bj;
      const float zf = acc[mt][2][e] + bff;
      const float zo = acc[mt][3][e] + bo;
      const float cp = c[idx];
      const float nc = cp * fsig(zf + 1.0f) + fsig(zi) * ftanh(zj);
      const float nh = ftanh(nc) * fsig(zo);
      c[idx] = nc;
      h_out[idx] = f2bf(nh);
      if (write_out) out_f32[idx] = nh;
    }
  }
}

// ---------------- host side ----------------

extern "C" void kernel_launch(void* const* d_in, const int* in_sizes, int n_in,
                              void* d_out, int out_size, void* d_ws, size_t ws_size,
                              hipStream_t stream) {
  const float* x   = (const float*)d_in[0];   // [512,256,128]
  const float* Wh  = (const float*)d_in[1];   // [128,1024]
  const float* bh  = (const float*)d_in[2];   // [1024]
  const float* k1  = (const float*)d_in[3];   // [2048,4096]
  const float* b1  = (const float*)d_in[4];   // [4096]
  const float* k2  = (const float*)d_in[5];   // [2048,4096]
  const float* b2  = (const float*)d_in[6];   // [4096]
  float* out = (float*)d_out;                 // [512,1024]

  char* ws = (char*)d_ws;
  size_t off = 0;
  auto alloc = [&](size_t bytes) -> void* {
    off = (off + 255) & ~(size_t)255;
    void* p = ws + off;
    off += bytes;
    return p;
  };

  unsigned short* k1T  = (unsigned short*)alloc((size_t)4096 * 2048 * 2);   // 16 MB
  unsigned short* k2T  = (unsigned short*)alloc((size_t)4096 * 2048 * 2);   // 16 MB
  unsigned short* WhT  = (unsigned short*)alloc((size_t)1024 * 128 * 2);
  unsigned short* Xbf  = (unsigned short*)alloc((size_t)B_DIM * T_DIM * NIN * 2);   // 33.5 MB
  unsigned short* proj = (unsigned short*)alloc((size_t)T_DIM * B_DIM * H_DIM * 2); // 256 MB
  unsigned short *h1[2], *h2[2];
  h1[0] = (unsigned short*)alloc((size_t)B_DIM * H_DIM * 2);
  h1[1] = (unsigned short*)alloc((size_t)B_DIM * H_DIM * 2);
  h2[0] = (unsigned short*)alloc((size_t)B_DIM * H_DIM * 2);
  h2[1] = (unsigned short*)alloc((size_t)B_DIM * H_DIM * 2);
  float* c1 = (float*)alloc((size_t)B_DIM * H_DIM * 4);
  float* c2 = (float*)alloc((size_t)B_DIM * H_DIM * 4);
  (void)ws_size; (void)in_sizes; (void)n_in; (void)out_size;

  // 1) weight conversion (fp32 -> bf16, transposed so K is contiguous per N-column)
  convT_kernel<<<dim3(4096 / 256, 2048), dim3(256), 0, stream>>>(k1, k1T, 2048, 4096);
  convT_kernel<<<dim3(4096 / 256, 2048), dim3(256), 0, stream>>>(k2, k2T, 2048, 4096);
  convT_kernel<<<dim3(1024 / 256, 128), dim3(256), 0, stream>>>(Wh, WhT, 128, 1024);
  {
    long nx = (long)B_DIM * T_DIM * NIN;
    conv_kernel<<<dim3((unsigned)((nx + 255) / 256)), dim3(256), 0, stream>>>(x, Xbf, nx);
  }

  // 2) zero initial state
  (void)hipMemsetAsync(h1[0], 0, (size_t)B_DIM * H_DIM * 2, stream);
  (void)hipMemsetAsync(h2[0], 0, (size_t)B_DIM * H_DIM * 2, stream);
  (void)hipMemsetAsync(c1, 0, (size_t)B_DIM * H_DIM * 4, stream);
  (void)hipMemsetAsync(c2, 0, (size_t)B_DIM * H_DIM * 4, stream);

  // 3) input projection (time-major, relu fused)
  proj_kernel<<<dim3(1024, 16), dim3(256), 0, stream>>>(Xbf, WhT, bh, proj);

  // 4) sequential scan: two fused WMMA step kernels per timestep
  for (int t = 0; t < T_DIM; ++t) {
    const int p = t & 1;
    const unsigned short* xt = proj + (size_t)t * B_DIM * H_DIM;
    lstm_step_kernel<<<dim3(8, 16), dim3(256), 0, stream>>>(
        xt, h1[p], k1T, b1, c1, h1[1 - p], nullptr, 0);
    lstm_step_kernel<<<dim3(8, 16), dim3(256), 0, stream>>>(
        h1[1 - p], h2[p], k2T, b2, c2, h2[1 - p], out, (t == T_DIM - 1) ? 1 : 0);
  }
}
